// TrafficGNN_12360915878436
// MI455X (gfx1250) — matrix operand order repaired and checked by reading, not compiled, not run on hardware
//
#include <hip/hip_runtime.h>
#include <hip/hip_bf16.h>
#include <math.h>

// ---------------------------------------------------------------------------
// Types for CDNA5 WMMA / TDM
// ---------------------------------------------------------------------------
typedef __attribute__((ext_vector_type(16))) __bf16   v16bf;
typedef __attribute__((ext_vector_type(8)))  float    v8f;
typedef __attribute__((ext_vector_type(4)))  unsigned v4u;
typedef __attribute__((ext_vector_type(8)))  int      v8i;
typedef __attribute__((ext_vector_type(4)))  int      v4i;

#define NEG_SLOPE 0.2f
#define EPS_F 1e-16f

// ---------------------------------------------------------------------------
// Helpers
// ---------------------------------------------------------------------------
__device__ __forceinline__ void split_bf16(float x, __bf16& hi, __bf16& lo) {
    hi = (__bf16)x;                  // RNE to bf16
    lo = (__bf16)(x - (float)hi);    // residual
}

// Monotone mapping float -> uint so that uint order == float order.
__device__ __forceinline__ unsigned enc_f(float f) {
    unsigned u = __float_as_uint(f);
    return (u & 0x80000000u) ? ~u : (u | 0x80000000u);
}
__device__ __forceinline__ float dec_f(unsigned u) {
    return (u & 0x80000000u) ? __uint_as_float(u & 0x7FFFFFFFu)
                             : __uint_as_float(~u);
}

__device__ __forceinline__ void atomic_add_f32(float* p, float v) {
    __hip_atomic_fetch_add(p, v, __ATOMIC_RELAXED, __HIP_MEMORY_SCOPE_AGENT);
}

// ---------------------------------------------------------------------------
// GEMM: Y[N,128] = X[N,K] @ W[K,128]   (K multiple of 32, K*128 <= 16384)
// W is staged into LDS once per block via the Tensor Data Mover (1-D tile),
// then each of the 8 waves computes a 16x16 tile with 3x error-compensated
// bf16 WMMA (hi/lo split, f32 accumulate).
// ---------------------------------------------------------------------------
__global__ __launch_bounds__(256) void gat_gemm_wmma(
    const float* __restrict__ X, const float* __restrict__ W,
    float* __restrict__ Y, int Nrows, int K)
{
    __shared__ float ldsW[16384];        // up to 128x128 f32 = 64 KB

    const int tid  = threadIdx.x;
    const int wave = tid >> 5;           // 0..7 -> column tile
    const int lane = tid & 31;
    const int half = lane >> 4;          // 0 or 1 (lane group)
    const int l15  = lane & 15;
    const int m0   = blockIdx.x << 4;    // row tile base
    const int n0   = wave << 4;          // col tile base
    const int nElems = K * 128;          // W element count (<= 16384)

    // ---- Stage W into LDS --------------------------------------------------
#if __has_builtin(__builtin_amdgcn_tensor_load_to_lds)
    if (wave == 0) {
        unsigned ldsAddr = (unsigned)(size_t)(void*)&ldsW[0]; // LDS byte offset
        unsigned long long ga = (unsigned long long)(size_t)W;
        // D# group 0: count=1 | lds_addr | global_addr[56:0] | type=2
        v4u g0 = { 1u,
                   ldsAddr,
                   (unsigned)(ga & 0xFFFFFFFFull),
                   (unsigned)((ga >> 32) & 0x01FFFFFFull) | 0x80000000u };
        // D# group 1: 1-D tensor, data_size=4B, tensor_dim0=tile_dim0=nElems
        v8i g1 = { (int)(2u << 16),                              // data_size=2 (4B)
                   (int)(((unsigned)nElems & 0xFFFFu) << 16),    // tensor_dim0[15:0]
                   (int)((((unsigned)nElems >> 16) & 0xFFFFu) | (1u << 16)), // dim0 hi | tensor_dim1=1
                   (int)(((unsigned)nElems & 0xFFFFu) << 16),    // tile_dim0 (fits 16b)
                   0,                                            // tile_dim1=0 (unused)
                   (int)(unsigned)nElems,                        // tensor_dim0_stride lo32
                   0, 0 };
        v4i gz = { 0, 0, 0, 0 };
#if defined(__clang_major__) && (__clang_major__ >= 23)
        v8i gz8 = { 0, 0, 0, 0, 0, 0, 0, 0 };
        __builtin_amdgcn_tensor_load_to_lds(g0, g1, gz, gz, gz8, 0);
#else
        __builtin_amdgcn_tensor_load_to_lds(g0, g1, gz, gz, 0);
#endif
        __builtin_amdgcn_s_wait_tensorcnt(0);
    }
#else
    for (int i = tid; i < nElems; i += 256) ldsW[i] = W[i];
#endif
    __syncthreads();

    const int arow   = m0 + l15;
    const int rowClp = arow < Nrows ? arow : (Nrows - 1);

    v8f acc = {};

    for (int kk = 0; kk < K; kk += 32) {
        v16bf ahi, alo, bhi, blo;

        // A fragment (16x32 bf16, ISA layout), from global:
        //  lane<16 : M=l15, elems 0..7 => K=kk+0..7,  elems 8..15 => K=kk+16..23
        //  lane>=16: M=l15, elems 0..7 => K=kk+8..15, elems 8..15 => K=kk+24..31
        const float* ap = X + (size_t)rowClp * K + kk + half * 8;
        #pragma unroll
        for (int e = 0; e < 8; ++e) {
            __bf16 h, l; split_bf16(ap[e], h, l);
            ahi[e] = h; alo[e] = l;
        }
        #pragma unroll
        for (int e = 0; e < 8; ++e) {
            __bf16 h, l; split_bf16(ap[16 + e], h, l);
            ahi[8 + e] = h; alo[8 + e] = l;
        }

        // B fragment (32x16 bf16, ISA layout), from LDS:
        //  lane<16 : N=l15, elems 0..7 => K=kk+0..7,  elems 8..15 => K=kk+16..23
        //  lane>=16: N=l15, elems 0..7 => K=kk+8..15, elems 8..15 => K=kk+24..31
        const float* bp = &ldsW[(kk + half * 8) * 128 + n0 + l15];
        #pragma unroll
        for (int e = 0; e < 16; ++e) {
            __bf16 h, l; split_bf16(bp[(e + (e >> 3) * 8) * 128], h, l);
            bhi[e] = h; blo[e] = l;
        }

        // Error-compensated bf16 product: hi*hi + hi*lo + lo*hi  (f32 accum)
        acc = __builtin_amdgcn_wmma_f32_16x16x32_bf16(false, ahi, false, bhi,
                                                      (short)0, acc, false, false);
        acc = __builtin_amdgcn_wmma_f32_16x16x32_bf16(false, ahi, false, blo,
                                                      (short)0, acc, false, false);
        acc = __builtin_amdgcn_wmma_f32_16x16x32_bf16(false, alo, false, bhi,
                                                      (short)0, acc, false, false);
    }

    // C/D layout: VGPR r: lanes 0-15 -> M=r, lanes 16-31 -> M=r+8; N=l15
    if (m0 + 16 <= Nrows) {            // fast path: whole tile in-bounds
        float* yp = Y + (size_t)(m0 + half * 8) * 128 + n0 + l15;
        #pragma unroll
        for (int r = 0; r < 8; ++r)
            yp[(size_t)r * 128] = acc[r];
    } else {
        #pragma unroll
        for (int r = 0; r < 8; ++r) {
            int row = m0 + r + half * 8;
            if (row < Nrows)
                Y[(size_t)row * 128 + n0 + l15] = acc[r];
        }
    }
}

// ---------------------------------------------------------------------------
// Per-(node,head) attention coefficients: a_src/a_dst = <h[n,h,:], att[h,:]>
// One wave (32 lanes) per (node, head); 64 channels -> 2 per lane.
// ---------------------------------------------------------------------------
__global__ __launch_bounds__(256) void gat_attn_coef(
    const float* __restrict__ Hf,
    const float* __restrict__ attS, const float* __restrict__ attD,
    float* __restrict__ aS, float* __restrict__ aD, int Nn)
{
    int unit = blockIdx.x * (blockDim.x >> 5) + (threadIdx.x >> 5);
    int lane = threadIdx.x & 31;
    int n = unit >> 1, h = unit & 1;
    if (n >= Nn) return;

    const float* hp = Hf + (size_t)n * 128 + h * 64;
    const float* sp = attS + h * 64;
    const float* dp = attD + h * 64;

    float s1 = hp[lane] * sp[lane] + hp[lane + 32] * sp[lane + 32];
    float s2 = hp[lane] * dp[lane] + hp[lane + 32] * dp[lane + 32];
    #pragma unroll
    for (int off = 16; off > 0; off >>= 1) {
        s1 += __shfl_down(s1, off, 32);
        s2 += __shfl_down(s2, off, 32);
    }
    if (lane == 0) { aS[n * 2 + h] = s1; aD[n * 2 + h] = s2; }
}

// ---------------------------------------------------------------------------
// Zero-fill n 32-bit words
// ---------------------------------------------------------------------------
__global__ __launch_bounds__(256) void gat_zero_u32(unsigned* __restrict__ p, long long n) {
    long long i = (long long)blockIdx.x * blockDim.x + threadIdx.x;
    if (i < n) p[i] = 0u;
}

// ---------------------------------------------------------------------------
// Edge pass 1: e = leaky_relu(a_src[src]+a_dst[dst]); store e; segment max.
// Edges [0,E) from edge_index, [E,Eprime) are self-loops.
// ---------------------------------------------------------------------------
__global__ __launch_bounds__(256) void gat_edge_scores(
    const int* __restrict__ srcA, const int* __restrict__ dstA,
    int E, int Eprime,
    const float* __restrict__ aS, const float* __restrict__ aD,
    float* __restrict__ ebuf, unsigned* __restrict__ mEnc)
{
    int i = blockIdx.x * blockDim.x + threadIdx.x;
    if (i >= Eprime) return;
    int s = (i < E) ? srcA[i] : (i - E);
    int d = (i < E) ? dstA[i] : (i - E);
    #pragma unroll
    for (int h = 0; h < 2; ++h) {
        float e = aS[s * 2 + h] + aD[d * 2 + h];
        e = (e > 0.f) ? e : NEG_SLOPE * e;
        ebuf[(size_t)i * 2 + h] = e;
        atomicMax(&mEnc[d * 2 + h], enc_f(e));
    }
}

// ---------------------------------------------------------------------------
// Edge pass 2: e <- exp(e - m[dst]); segment sum.
// ---------------------------------------------------------------------------
__global__ __launch_bounds__(256) void gat_edge_exp(
    const int* __restrict__ dstA, int E, int Eprime,
    const unsigned* __restrict__ mEnc,
    float* __restrict__ ebuf, float* __restrict__ ssum)
{
    int i = blockIdx.x * blockDim.x + threadIdx.x;
    if (i >= Eprime) return;
    int d = (i < E) ? dstA[i] : (i - E);
    #pragma unroll
    for (int h = 0; h < 2; ++h) {
        float e  = ebuf[(size_t)i * 2 + h];
        float mv = dec_f(mEnc[d * 2 + h]);
        float ex = expf(e - mv);
        ebuf[(size_t)i * 2 + h] = ex;
        atomic_add_f32(&ssum[d * 2 + h], ex);
    }
}

// ---------------------------------------------------------------------------
// Edge pass 3 (heavy): acc[dst, j] += alpha * h[src, j]  for j in [0,128).
// 128 lanes per edge -> coalesced 512B gather + coalesced f32 atomics.
// ---------------------------------------------------------------------------
__global__ __launch_bounds__(256) void gat_edge_scatter(
    const int* __restrict__ srcA, const int* __restrict__ dstA,
    int E, int Eprime,
    const float* __restrict__ ebuf, const float* __restrict__ ssum,
    const float* __restrict__ Hf, float* __restrict__ acc)
{
    int eidx = blockIdx.x * 2 + (threadIdx.x >> 7);
    int j    = threadIdx.x & 127;          // channel (h*64+c)
    if (eidx >= Eprime) return;
    int s = (eidx < E) ? srcA[eidx] : (eidx - E);
    int d = (eidx < E) ? dstA[eidx] : (eidx - E);
    int h = j >> 6;
    float alpha = ebuf[(size_t)eidx * 2 + h] / (ssum[d * 2 + h] + EPS_F);
    atomic_add_f32(&acc[(size_t)d * 128 + j], alpha * Hf[(size_t)s * 128 + j]);
}

// ---------------------------------------------------------------------------
// Head-mean + bias + relu : x2[n,c] = relu(0.5*(acc[n,c]+acc[n,64+c]) + b[c])
// ---------------------------------------------------------------------------
__global__ __launch_bounds__(256) void gat_head_mean_relu(
    const float* __restrict__ acc, const float* __restrict__ bias,
    float* __restrict__ out64, int Nn)
{
    int i = blockIdx.x * blockDim.x + threadIdx.x;
    if (i >= Nn * 64) return;
    int n = i >> 6, c = i & 63;
    float v = 0.5f * (acc[(size_t)n * 128 + c] + acc[(size_t)n * 128 + 64 + c]) + bias[c];
    out64[i] = (v > 0.f) ? v : 0.f;
}

// ---------------------------------------------------------------------------
// Final FC: out[n] = <x2[n,:], fcW> + fcb   (one wave per node)
// ---------------------------------------------------------------------------
__global__ __launch_bounds__(256) void gat_fc(
    const float* __restrict__ x64, const float* __restrict__ fcW,
    const float* __restrict__ fcB, float* __restrict__ out, int Nn)
{
    int n    = blockIdx.x * (blockDim.x >> 5) + (threadIdx.x >> 5);
    int lane = threadIdx.x & 31;
    if (n >= Nn) return;
    float s = x64[(size_t)n * 64 + lane] * fcW[lane]
            + x64[(size_t)n * 64 + 32 + lane] * fcW[32 + lane];
    #pragma unroll
    for (int off = 16; off > 0; off >>= 1) s += __shfl_down(s, off, 32);
    if (lane == 0) out[n] = s + fcB[0];
}

// ---------------------------------------------------------------------------
// Host orchestration
// ---------------------------------------------------------------------------
extern "C" void kernel_launch(void* const* d_in, const int* in_sizes, int n_in,
                              void* d_out, int out_size, void* d_ws, size_t ws_size,
                              hipStream_t stream)
{
    const int F_IN = 128, HC = 128, C = 64;
    const int N = in_sizes[0] / F_IN;
    const int E = in_sizes[1] / 2;
    const int Eprime = E + N;

    const float* x     = (const float*)d_in[0];
    const int*   srcA  = (const int*)d_in[1];          // edge_index[0]
    const int*   dstA  = ((const int*)d_in[1]) + E;    // edge_index[1]
    const float* W1    = (const float*)d_in[2];
    const float* aS1   = (const float*)d_in[3];
    const float* aD1   = (const float*)d_in[4];
    const float* b1    = (const float*)d_in[5];
    const float* W2    = (const float*)d_in[6];
    const float* aS2   = (const float*)d_in[7];
    const float* aD2   = (const float*)d_in[8];
    const float* b2    = (const float*)d_in[9];
    const float* fcW   = (const float*)d_in[10];
    const float* fcB   = (const float*)d_in[11];
    float*       out   = (float*)d_out;

    // Workspace carve-up (256B aligned)
    size_t off = 0;
    auto carve = [&](size_t bytes) {
        off = (off + 255) & ~(size_t)255;
        void* p = (char*)d_ws + off;
        off += bytes;
        return p;
    };
    float*    hbuf = (float*)   carve((size_t)N * HC * 4);
    float*    acc  = (float*)   carve((size_t)N * HC * 4);
    float*    x2   = (float*)   carve((size_t)N * C  * 4);
    float*    aS   = (float*)   carve((size_t)N * 2  * 4);
    float*    aD   = (float*)   carve((size_t)N * 2  * 4);
    unsigned* mEnc = (unsigned*)carve((size_t)N * 2  * 4);
    float*    ssum = (float*)   carve((size_t)N * 2  * 4);
    float*    ebuf = (float*)   carve((size_t)Eprime * 2 * 4);
    (void)ws_size;

    const int TB = 256;
    dim3 blk(TB);
    int gemmGrid  = (N + 15) / 16;
    int coefGrid  = (N * 2 + 7) / 8;       // 8 (node,head) units per block
    int edgeGrid  = (Eprime + TB - 1) / TB;
    int scatGrid  = (Eprime + 1) / 2;      // 2 edges per 256-thread block
    int meanGrid  = (N * 64 + TB - 1) / TB;
    int fcGrid    = (N + 7) / 8;           // 8 nodes per block
    long long nAcc = (long long)N * HC;
    long long nNH  = (long long)N * 2;
    int zAccGrid = (int)((nAcc + TB - 1) / TB);
    int zNHGrid  = (int)((nNH + TB - 1) / TB);

    // ---------------- Layer 1 ----------------
    gat_gemm_wmma<<<gemmGrid, blk, 0, stream>>>(x, W1, hbuf, N, F_IN);
    gat_attn_coef<<<coefGrid, blk, 0, stream>>>(hbuf, aS1, aD1, aS, aD, N);
    gat_zero_u32<<<zAccGrid, blk, 0, stream>>>((unsigned*)acc, nAcc);
    gat_zero_u32<<<zNHGrid,  blk, 0, stream>>>(mEnc, nNH);
    gat_zero_u32<<<zNHGrid,  blk, 0, stream>>>((unsigned*)ssum, nNH);
    gat_edge_scores<<<edgeGrid, blk, 0, stream>>>(srcA, dstA, E, Eprime, aS, aD, ebuf, mEnc);
    gat_edge_exp<<<edgeGrid, blk, 0, stream>>>(dstA, E, Eprime, mEnc, ebuf, ssum);
    gat_edge_scatter<<<scatGrid, blk, 0, stream>>>(srcA, dstA, E, Eprime, ebuf, ssum, hbuf, acc);
    gat_head_mean_relu<<<meanGrid, blk, 0, stream>>>(acc, b1, x2, N);

    // ---------------- Layer 2 ----------------
    gat_gemm_wmma<<<gemmGrid, blk, 0, stream>>>(x2, W2, hbuf, N, C);
    gat_attn_coef<<<coefGrid, blk, 0, stream>>>(hbuf, aS2, aD2, aS, aD, N);
    gat_zero_u32<<<zAccGrid, blk, 0, stream>>>((unsigned*)acc, nAcc);
    gat_zero_u32<<<zNHGrid,  blk, 0, stream>>>(mEnc, nNH);
    gat_zero_u32<<<zNHGrid,  blk, 0, stream>>>((unsigned*)ssum, nNH);
    gat_edge_scores<<<edgeGrid, blk, 0, stream>>>(srcA, dstA, E, Eprime, aS, aD, ebuf, mEnc);
    gat_edge_exp<<<edgeGrid, blk, 0, stream>>>(dstA, E, Eprime, mEnc, ebuf, ssum);
    gat_edge_scatter<<<scatGrid, blk, 0, stream>>>(srcA, dstA, E, Eprime, ebuf, ssum, hbuf, acc);
    gat_head_mean_relu<<<meanGrid, blk, 0, stream>>>(acc, b2, x2, N);

    // ---------------- Readout ----------------
    gat_fc<<<fcGrid, blk, 0, stream>>>(x2, fcW, fcB, out, N);
}
